// GNBlock_36352603193748
// MI455X (gfx1250) — matrix operand order repaired and checked by reading, not compiled, not run on hardware
//
#include <hip/hip_runtime.h>

// ---------------------------------------------------------------------------
// GN block for MI455X (gfx1250, wave32, WMMA + TDM).
// f16 WMMA (16x16x32, f32 accum) tile MLPs. 256 rows/WG, 2 M-tiles per wave
// (each LDS B fragment feeds two WMMAs). Activations updated in place.
// Layer weights are DMA'd into LDS by the Tensor Data Mover
// (tensor_load_to_lds, TENSORcnt), double-buffered in two 64KB half-layer
// chunks so the DMA overlaps WMMA compute and the input gather.
// LDS: act 128KB + weights 2x64KB + psum 1KB = 257KB (< 320KB/WGP).
// Round 7: software-pipeline the B-fragment LDS loads one K-chunk ahead so
// the per-pair s_wait_dscnt 0x0 stall (round 6 asm) becomes a non-blocking
// s_wait_dscnt 0x2 behind two WMMAs.
// ---------------------------------------------------------------------------

typedef _Float16 half_t;
typedef __attribute__((ext_vector_type(16))) _Float16 v16h;
typedef __attribute__((ext_vector_type(8)))  _Float16 v8h;
typedef __attribute__((ext_vector_type(4)))  _Float16 v4h;
typedef __attribute__((ext_vector_type(8)))  float    v8f;
typedef __attribute__((ext_vector_type(4)))  float    v4f;
typedef __attribute__((ext_vector_type(4)))  unsigned int u32x4;
typedef __attribute__((ext_vector_type(8)))  int      i32x8;
typedef __attribute__((ext_vector_type(4)))  int      i32x4;

#define THREADS 256   // 8 wave32 per workgroup
#define MTILE   256   // rows per workgroup (2 x 16-row tiles per wave)
#define HID     256   // activation row stride (fixed for all layers)

#define N_NODES 50000
#define N_EDGES 100000
#define BS      4
#define EROWS   ((long long)N_EDGES * BS)   // 400000
#define NROWS   ((long long)N_NODES * BS)   // 200000

#if !defined(__HIP_DEVICE_COMPILE__)
// Host pass: amdgcn builtins are aux-target; __has_builtin can't see them,
// but Sema resolves the calls. Device pass does the real check below.
#define USE_TDM 1
#elif __has_builtin(__builtin_amdgcn_tensor_load_to_lds) && \
      __has_builtin(__builtin_amdgcn_s_wait_tensorcnt)
#define USE_TDM 1
#else
#error "TDM builtins unavailable on this toolchain (device pass)."
#endif

struct MlpDesc {
    const half_t* w[4];   // transposed f16 weights: Wt[out][in]
    const float*  b[4];   // f32 biases (from d_in directly)
};

// ---------------------------------------------------------------------------
// Issue one TDM transfer: contiguous nbytes from global src -> LDS dst.
// 1-D descriptor: data_size=8B, tile_dim0 = tensor_dim0 = nbytes/8 (<= 8192
// for our 64KB chunks, fits the 16-bit tile_dim0). Caller: one wave only.
// ---------------------------------------------------------------------------
__device__ __forceinline__ void tdm_issue(half_t* dst, const half_t* src, int nbytes)
{
    const unsigned lds = (unsigned)(unsigned long long)dst;     // LDS byte addr
    const unsigned long long ga = (unsigned long long)src;
    const unsigned n8 = ((unsigned)nbytes) >> 3;                // 8B elements

    u32x4 g0;
    g0[0] = 1u;                                            // count=1 (user D#)
    g0[1] = lds;                                           // lds_addr
    g0[2] = (unsigned)ga;                                  // global_addr[31:0]
    g0[3] = (unsigned)((ga >> 32) & 0x01FFFFFFull)         // global_addr[56:32]
          | (2u << 30);                                    // type=2 ("image")

    i32x8 g1;
    g1[0] = (int)(3u << 16);                               // data_size=3 (8B)
    g1[1] = (int)((n8 & 0xFFFFu) << 16);                   // tensor_dim0 lo16
    g1[2] = (int)((n8 >> 16) | (1u << 16));                // dim0 hi16 | tensor_dim1=1
    g1[3] = (int)((n8 & 0xFFFFu) << 16);                   // tile_dim0
    g1[4] = 0;                                             // tile_dim1/2 unused
    g1[5] = (int)n8;                                       // tensor_dim0_stride
    g1[6] = 0;
    g1[7] = 0;

    const i32x4 gz4 = {0, 0, 0, 0};                        // groups 2/3 unused
    const i32x8 gz8 = {0, 0, 0, 0, 0, 0, 0, 0};            // trailing group (6-arg form)
    __builtin_amdgcn_tensor_load_to_lds(g0, g1, gz4, gz4, gz8, 0);
}

// ---------------------------------------------------------------------------
// Compute one column-half of a layer: nt tiles [HALF*NOUT/32, ...).
// afrag: this wave's 32 rows (2 tiles x KC chunks), preloaded in VGPRs.
// whalf: K x NOUT/2 f16 transposed weights for this half's columns.
// B LDS loads are software-pipelined one K-chunk ahead of the WMMAs.
// ---------------------------------------------------------------------------
template<int K, int NOUT, bool LAST, int HALF, int KC>
__device__ __forceinline__ void compute_half(
    const v16h (&afrag)[2][KC], half_t* __restrict__ act,
    const half_t* __restrict__ whalf,
    float* __restrict__ psum, const float* __restrict__ bias,
    float* __restrict__ out, long long rowBase, long long Mtotal, bool doAccum,
    int mrow0, int g, int ln)
{
    constexpr int NTH = NOUT / 32;    // 16-col tiles per half

    for (int ntl = 0; ntl < NTH; ++ntl) {
        const int lc   = ntl * 16 + ln;            // col within this half
        const int ncol = HALF * (NOUT / 2) + lc;   // global output col
        // B layout (16-bit 32x16): lane holds col N=ln; halves 0..7 -> K=32kc+8g+0..7, 8..15 -> +16.
        const half_t* bcol = whalf + (size_t)lc * K + 8 * g;

        const float bv = bias[ncol];   // hoisted: overlaps the MMA chain

        v8f acc0 = {};
        v8f acc1 = {};
        // Prologue: first B pair in flight before the loop.
        v8h blo = *(const v8h*)(bcol);
        v8h bhi = *(const v8h*)(bcol + 16);
        #pragma unroll
        for (int kc = 0; kc < KC; ++kc) {
            v16h bf = __builtin_shufflevector(blo, bhi,
                0,1,2,3,4,5,6,7,8,9,10,11,12,13,14,15);
            if (kc + 1 < KC) {          // prefetch next chunk before the WMMAs
                blo = *(const v8h*)(bcol + 32 * (kc + 1));
                bhi = *(const v8h*)(bcol + 32 * (kc + 1) + 16);
            }
            // One B fragment feeds both M-tiles (2x LDS reuse).
            acc0 = __builtin_amdgcn_wmma_f32_16x16x32_f16(
                     false, afrag[0][kc], false, bf, (short)0, acc0, false, false);
            acc1 = __builtin_amdgcn_wmma_f32_16x16x32_f16(
                     false, afrag[1][kc], false, bf, (short)0, acc1, false, false);
        }

        if (!LAST) {
            // C/D layout: VGPR r -> row M = r + 8g (+16 per tile), col N = ln.
            #pragma unroll
            for (int r = 0; r < 8; ++r) {
                float v0 = acc0[r] + bv;
                v0 = v0 > 0.0f ? v0 : 0.0f;          // ReLU
                act[(size_t)(mrow0 + r + 8 * g) * HID + ncol] = (half_t)v0;
                float v1 = acc1[r] + bv;
                v1 = v1 > 0.0f ? v1 : 0.0f;
                act[(size_t)(mrow0 + 16 + r + 8 * g) * HID + ncol] = (half_t)v1;
            }
        } else {
            #pragma unroll
            for (int r = 0; r < 8; ++r) {
                #pragma unroll
                for (int t = 0; t < 2; ++t) {
                    const float v = (t ? acc1[r] : acc0[r]) + bv;  // no ReLU last
                    const long long row = rowBase + mrow0 + 16 * t + r + 8 * g;
                    if (row < Mtotal) {
                        out[row * 64 + ncol] = v;
                        if (doAccum)
                            atomicAdd(&psum[(int)(row & 3) * 64 + ncol], v);
                    }
                }
            }
        }
    }
}

// ---------------------------------------------------------------------------
// One MLP layer (in place on act), weights double-buffered per column-half.
// On entry: half0 weights in bufA (<=1 TDM outstanding past it), half1 chunk
// in flight into bufB. After each half: issue the next pipeline chunk.
// ---------------------------------------------------------------------------
template<int K, int NOUT, bool LAST>
__device__ __forceinline__ void mlp_layer_pipe(
    half_t* __restrict__ act, float* __restrict__ psum,
    const float* __restrict__ bias, float* __restrict__ out,
    long long rowBase, long long Mtotal, bool doAccum,
    const half_t* bufA, const half_t* bufB,
    const half_t* next0, int next0Bytes, half_t* next0Buf,
    const half_t* next1, int next1Bytes, half_t* next1Buf)
{
    constexpr int KC = K / 32;

    const int tid  = threadIdx.x;
    const int wave = tid >> 5;
    const int lane = tid & 31;
    const int g    = lane >> 4;
    const int ln   = lane & 15;
    const int mrow0 = wave * 32;

    // A fragments once per layer (both halves share them); must be read
    // before the in-place writes below. A layout (16-bit 16x32): lane holds
    // row M=ln; halves 0..7 -> K=32kc+8g+0..7, 8..15 -> +16.
    v16h afrag[2][KC];
    #pragma unroll
    for (int t = 0; t < 2; ++t) {
        const half_t* arow = act + (size_t)(mrow0 + 16 * t + ln) * HID + 8 * g;
        #pragma unroll
        for (int kc = 0; kc < KC; ++kc) {
            v8h alo = *(const v8h*)(arow + 32 * kc);
            v8h ahi = *(const v8h*)(arow + 32 * kc + 16);
            afrag[t][kc] = __builtin_shufflevector(alo, ahi,
                0,1,2,3,4,5,6,7,8,9,10,11,12,13,14,15);
        }
    }

    if (wave == 0) __builtin_amdgcn_s_wait_tensorcnt((short)1);
    __syncthreads();                    // half0 weights visible to all waves
    compute_half<K, NOUT, LAST, 0, KC>(afrag, act, bufA, psum, bias, out,
                                       rowBase, Mtotal, doAccum, mrow0, g, ln);
    __syncthreads();                    // all waves done reading bufA
    if (wave == 0) {
        if (next0) tdm_issue(next0Buf, next0, next0Bytes);
        __builtin_amdgcn_s_wait_tensorcnt(LAST ? (short)0 : (short)1);
    }
    __syncthreads();                    // half1 weights visible
    compute_half<K, NOUT, LAST, 1, KC>(afrag, act, bufB, psum, bias, out,
                                       rowBase, Mtotal, doAccum, mrow0, g, ln);
    __syncthreads();                    // all waves done reading bufB
    if (wave == 0 && next1) tdm_issue(next1Buf, next1, next1Bytes);
}

// ---------------------------------------------------------------------------
// 4-layer MLP pipeline. Chunk schedule (issued two ahead):
//   s0 l0h0, s1 l0h1 (issued by the kernel before input assembly),
//   s2 l1h0, s3 l1h1, s4 l2h0, s5 l2h1, s6 l3h0, s7 l3h1.
// ---------------------------------------------------------------------------
template<int K0>
__device__ __forceinline__ void run_mlp_tile(
    half_t* __restrict__ act, half_t* __restrict__ wbuf0, half_t* __restrict__ wbuf1,
    float* __restrict__ psum, const MlpDesc md, float* __restrict__ out,
    long long rowBase, long long Mtotal, bool doAccum)
{
    mlp_layer_pipe<K0, HID, false>(act, psum, md.b[0], out, rowBase, Mtotal, doAccum,
        wbuf0, wbuf1,
        md.w[1],             128 * HID * 2, wbuf0,
        md.w[1] + 128 * HID, 128 * HID * 2, wbuf1);
    mlp_layer_pipe<HID, HID, false>(act, psum, md.b[1], out, rowBase, Mtotal, doAccum,
        wbuf0, wbuf1,
        md.w[2],             128 * HID * 2, wbuf0,
        md.w[2] + 128 * HID, 128 * HID * 2, wbuf1);
    mlp_layer_pipe<HID, HID, false>(act, psum, md.b[2], out, rowBase, Mtotal, doAccum,
        wbuf0, wbuf1,
        md.w[3],            32 * HID * 2, wbuf0,
        md.w[3] + 32 * HID, 32 * HID * 2, wbuf1);
    mlp_layer_pipe<HID, 64, true>(act, psum, md.b[3], out, rowBase, Mtotal, doAccum,
        wbuf0, wbuf1,
        nullptr, 0, nullptr,
        nullptr, 0, nullptr);
}

// ---------------------------------------------------------------------------
// Weight convert: W[k][n] (f32) -> Wt[n][k] (f16), contiguous in k.
// ---------------------------------------------------------------------------
__global__ void wconv_kernel(const float* __restrict__ src,
                             half_t* __restrict__ dst, int K, int N)
{
    const int idx = blockIdx.x * blockDim.x + threadIdx.x;
    if (idx < K * N) {
        const int o = idx / K;
        const int k = idx - o * K;
        dst[idx] = (half_t)src[(size_t)k * N + o];
    }
}

__global__ void zero_kernel(float* p, int n)
{
    const int idx = blockIdx.x * blockDim.x + threadIdx.x;
    if (idx < n) p[idx] = 0.0f;
}

// ---------------------------------------------------------------------------
// Edge block: rows = e*4+b. Input = [g(b) | n(senders[e],b) | n(receivers[e],b) | e(row)]
// ---------------------------------------------------------------------------
__global__ void __launch_bounds__(THREADS)
edge_mlp_kernel(const float* __restrict__ graph, const float* __restrict__ nodef,
                const float* __restrict__ edgef,
                const int* __restrict__ senders, const int* __restrict__ recvs,
                MlpDesc md, float* __restrict__ out_e, float* __restrict__ esum_accum)
{
    extern __shared__ __align__(16) char smem[];
    half_t* act   = (half_t*)smem;                   // 256*256 f16 (128KB)
    half_t* wbuf0 = act + MTILE * HID;               // 64KB half-layer chunk
    half_t* wbuf1 = wbuf0 + 128 * HID;               // 64KB half-layer chunk
    float*  psum  = (float*)(wbuf1 + 128 * HID);     // 4*64 f32

    const int tid = threadIdx.x;
    psum[tid] = 0.0f;

    // Kick off the weight DMA pipeline before the gather (overlaps both).
    if ((tid >> 5) == 0) {
        tdm_issue(wbuf0, md.w[0],             128 * 256 * 2);
        tdm_issue(wbuf1, md.w[0] + 128 * 256, 128 * 256 * 2);
    }

    const long long rowBase = (long long)blockIdx.x * MTILE;

    // Assemble f16 inputs: 256 rows x 64 float4 segments (zeros past EROWS).
    for (int idx = tid; idx < MTILE * 64; idx += THREADS) {
        const int i = idx >> 6;
        const int q = idx & 63;
        const long long row = rowBase + i;
        v4h h = {};
        if (row < EROWS) {
            const int e  = (int)(row >> 2);
            const int b  = (int)(row & 3);
            const int c4 = (q & 15) * 4;
            const float* src;
            if (q < 16)      src = graph + b * 64 + c4;
            else if (q < 32) src = nodef + ((size_t)senders[e] * 4 + b) * 64 + c4;
            else if (q < 48) src = nodef + ((size_t)recvs[e]   * 4 + b) * 64 + c4;
            else             src = edgef + (size_t)row * 64 + c4;
            const v4f v = *(const v4f*)src;
            h = (v4h){ (half_t)v[0], (half_t)v[1], (half_t)v[2], (half_t)v[3] };
        }
        *(v4h*)(act + (size_t)i * HID + q * 4) = h;
    }
    __syncthreads();

    run_mlp_tile<256>(act, wbuf0, wbuf1, psum, md, out_e, rowBase, EROWS, true);

    // psum complete (layer ends with a barrier): fold into global edge_sum.
    atomicAdd(&esum_accum[tid], psum[tid]);
}

// ---------------------------------------------------------------------------
// Node block: rows = n*4+b. Input = [g(b) | n(row) | edge_sum(b)], K0 = 192.
// ---------------------------------------------------------------------------
__global__ void __launch_bounds__(THREADS)
node_mlp_kernel(const float* __restrict__ graph, const float* __restrict__ nodef,
                const float* __restrict__ esum,
                MlpDesc md, float* __restrict__ out_n, float* __restrict__ nagg_accum)
{
    extern __shared__ __align__(16) char smem[];
    half_t* act   = (half_t*)smem;
    half_t* wbuf0 = act + MTILE * HID;
    half_t* wbuf1 = wbuf0 + 128 * HID;
    float*  psum  = (float*)(wbuf1 + 128 * HID);

    const int tid = threadIdx.x;
    psum[tid] = 0.0f;

    if ((tid >> 5) == 0) {
        tdm_issue(wbuf0, md.w[0],             128 * 192 * 2);
        tdm_issue(wbuf1, md.w[0] + 128 * 192, 128 * 192 * 2);
    }

    const long long rowBase = (long long)blockIdx.x * MTILE;

    // 256 rows x 48 float4 segments (192 input cols; act stride stays 256).
    for (int idx = tid; idx < MTILE * 48; idx += THREADS) {
        const int i = idx / 48;
        const int q = idx - i * 48;
        const int c4 = (q & 15) * 4;
        const long long row = rowBase + i;
        v4h h = {};
        if (row < NROWS) {
            const int b = (int)(row & 3);
            const float* src;
            if (q < 16)      src = graph + b * 64 + c4;
            else if (q < 32) src = nodef + (size_t)row * 64 + c4;
            else             src = esum + b * 64 + c4;
            const v4f v = *(const v4f*)src;
            h = (v4h){ (half_t)v[0], (half_t)v[1], (half_t)v[2], (half_t)v[3] };
        }
        *(v4h*)(act + (size_t)i * HID + q * 4) = h;
    }
    __syncthreads();

    run_mlp_tile<192>(act, wbuf0, wbuf1, psum, md, out_n, rowBase, NROWS, true);

    atomicAdd(&nagg_accum[tid], psum[tid]);
}

// ---------------------------------------------------------------------------
// Graph block: 4 rows. Input = [g(b) | n_agg(b) | edge_sum(b)], K0 = 192.
// ---------------------------------------------------------------------------
__global__ void __launch_bounds__(THREADS)
graph_mlp_kernel(const float* __restrict__ graph, const float* __restrict__ esum,
                 const float* __restrict__ nagg,
                 MlpDesc md, float* __restrict__ out_g)
{
    extern __shared__ __align__(16) char smem[];
    half_t* act   = (half_t*)smem;
    half_t* wbuf0 = act + MTILE * HID;
    half_t* wbuf1 = wbuf0 + 128 * HID;
    float*  psum  = (float*)(wbuf1 + 128 * HID);

    const int tid = threadIdx.x;
    psum[tid] = 0.0f;

    if ((tid >> 5) == 0) {
        tdm_issue(wbuf0, md.w[0],             128 * 192 * 2);
        tdm_issue(wbuf1, md.w[0] + 128 * 192, 128 * 192 * 2);
    }

    for (int idx = tid; idx < MTILE * 48; idx += THREADS) {
        const int i = idx / 48;
        const int q = idx - i * 48;
        const int c4 = (q & 15) * 4;
        v4h h = {};
        if (i < BS) {
            const float* src;
            if (q < 16)      src = graph + i * 64 + c4;
            else if (q < 32) src = nagg + i * 64 + c4;
            else             src = esum + i * 64 + c4;
            const v4f v = *(const v4f*)src;
            h = (v4h){ (half_t)v[0], (half_t)v[1], (half_t)v[2], (half_t)v[3] };
        }
        *(v4h*)(act + (size_t)i * HID + q * 4) = h;
    }
    __syncthreads();

    run_mlp_tile<192>(act, wbuf0, wbuf1, psum, md, out_g, 0, BS, false);
}

// ---------------------------------------------------------------------------
// Host launcher.
// d_in: 0 graph_feat, 1 node_feats, 2 edge_feats, 3 senders, 4 receivers,
//       5..12 edge (W,b)x4, 13..20 node (W,b)x4, 21..28 graph (W,b)x4.
// d_out: [g_new 4*64 | n_new 200000*64 | e_new 400000*64] f32.
// ---------------------------------------------------------------------------
extern "C" void kernel_launch(void* const* d_in, const int* in_sizes, int n_in,
                              void* d_out, int out_size, void* d_ws, size_t ws_size,
                              hipStream_t stream)
{
    (void)in_sizes; (void)n_in; (void)out_size; (void)ws_size;

    const float* graph   = (const float*)d_in[0];
    const float* nodef   = (const float*)d_in[1];
    const float* edgef   = (const float*)d_in[2];
    const int*   senders = (const int*)d_in[3];
    const int*   recvs   = (const int*)d_in[4];

    static const int eK[4] = {256, 256, 256, 256}, eN[4] = {256, 256, 256, 64};
    static const int nK[4] = {192, 256, 256, 256}, nN[4] = {256, 256, 256, 64};

    half_t* wp = (half_t*)d_ws;
    size_t off = 0;
    MlpDesc emd, nmd, gmd;
    half_t* wdst[12];

    for (int l = 0; l < 4; ++l) {
        wdst[l] = wp + off; emd.w[l] = wdst[l];
        emd.b[l] = (const float*)d_in[6 + 2 * l];
        off += (size_t)eN[l] * eK[l];
    }
    for (int l = 0; l < 4; ++l) {
        wdst[4 + l] = wp + off; nmd.w[l] = wdst[4 + l];
        nmd.b[l] = (const float*)d_in[14 + 2 * l];
        off += (size_t)nN[l] * nK[l];
    }
    for (int l = 0; l < 4; ++l) {
        wdst[8 + l] = wp + off; gmd.w[l] = wdst[8 + l];
        gmd.b[l] = (const float*)d_in[22 + 2 * l];
        off += (size_t)nN[l] * nK[l];
    }

    float* esum = (float*)(wp + off);   // 4*64
    float* nagg = esum + 256;           // 4*64

    // 1) convert + transpose weights to f16
    for (int l = 0; l < 4; ++l) {
        int tot = eK[l] * eN[l];
        wconv_kernel<<<(tot + 255) / 256, 256, 0, stream>>>(
            (const float*)d_in[5 + 2 * l], wdst[l], eK[l], eN[l]);
    }
    for (int l = 0; l < 4; ++l) {
        int tot = nK[l] * nN[l];
        wconv_kernel<<<(tot + 255) / 256, 256, 0, stream>>>(
            (const float*)d_in[13 + 2 * l], wdst[4 + l], nK[l], nN[l]);
        wconv_kernel<<<(tot + 255) / 256, 256, 0, stream>>>(
            (const float*)d_in[21 + 2 * l], wdst[8 + l], nK[l], nN[l]);
    }

    // 2) zero the two global accumulators (edge_sum, n_agg)
    zero_kernel<<<2, 256, 0, stream>>>(esum, 512);

    float* out_g = (float*)d_out;
    float* out_n = out_g + 256;
    float* out_e = out_n + (size_t)NROWS * 64;

    const size_t smem_bytes = (size_t)(MTILE * HID * sizeof(half_t))    // act
                            + (size_t)(HID * HID * sizeof(half_t))     // wbuf0+1
                            + 256 * sizeof(float);                     // psum

    // 3) edge block (1563 WGs x 256 rows, last partially masked)
    edge_mlp_kernel<<<(int)((EROWS + MTILE - 1) / MTILE), THREADS, smem_bytes, stream>>>(
        graph, nodef, edgef, senders, recvs, emd, out_e, esum);

    // 4) node block (782 WGs)
    node_mlp_kernel<<<(int)((NROWS + MTILE - 1) / MTILE), THREADS, smem_bytes, stream>>>(
        graph, nodef, esum, nmd, out_n, nagg);

    // 5) graph block (1 WG, 4 rows)
    graph_mlp_kernel<<<1, THREADS, smem_bytes, stream>>>(
        graph, esum, nagg, gmd, out_g);
}
